// ChannelDropout_43293270343996
// MI455X (gfx1250) — matrix operand order repaired and checked by reading, not compiled, not run on hardware
//
#include <hip/hip_runtime.h>
#include <hip/hip_bf16.h>
#include <math.h>

// ---------------------------------------------------------------------------
// Problem constants (match reference): x[32,512,56,56] fp32
// ---------------------------------------------------------------------------
#define BB   32
#define CC   512
#define HW   3136          // 56*56
#define HID  32            // C / RED
#define KSEL 256           // C/2, top-k count

typedef __attribute__((ext_vector_type(16))) _Float16 v16h;
typedef __attribute__((ext_vector_type(8)))  float    v8f;

// ---------------------------------------------------------------------------
// Kernel 1: per-(b,c) spatial mean + max.  One wave32 per row of 3136 floats.
// 3136 = 49 * 64 -> each lane reads 49 float2 (fully coalesced 256B/wave/op).
// ---------------------------------------------------------------------------
__global__ void k_reduce_meanmax(const float* __restrict__ x,
                                 float* __restrict__ yws) {
    const int tid  = threadIdx.x;
    const int lane = tid & 31;
    const int row  = (blockIdx.x * blockDim.x + tid) >> 5;   // 0 .. B*C-1
    if (row >= BB * CC) return;

    const float2* __restrict__ p = (const float2*)(x + (size_t)row * HW);
    float s  = 0.0f;
    float mx = -3.402823466e38f;
#pragma unroll 7
    for (int i = 0; i < 49; ++i) {
        float2 v = p[lane + i * 32];
        s += v.x + v.y;
        mx = fmaxf(mx, fmaxf(v.x, v.y));
    }
    // wave32 reduction
#pragma unroll
    for (int off = 16; off > 0; off >>= 1) {
        s  += __shfl_xor(s, off, 32);
        mx  = fmaxf(mx, __shfl_xor(mx, off, 32));
    }
    if (lane == 0)
        yws[row] = s * (1.0f / (float)HW) + mx;
}

// ---------------------------------------------------------------------------
// Kernel 2: fused MLP (WMMA) + sigmoid + top-k threshold + random gating.
// Grid = B/16 blocks, 512 threads (16 waves).  Batch tile M = 16.
//
// GEMM1: y[16,512] @ w1[512,32]  -> 2 N-tiles, 16 K-steps of 16x16x32 (f16)
// GEMM2: h[16,32]  @ w2[32,512]  -> 32 N-tiles, 1 K-step each
// ---------------------------------------------------------------------------
__global__ __launch_bounds__(512)
void k_mlp_topk(const float* __restrict__ yws,
                const float* __restrict__ w1,   // [512,32] row-major
                const float* __restrict__ b1,   // [32]
                const float* __restrict__ pa,   // scalar PReLU slope
                const float* __restrict__ w2,   // [32,512] row-major
                const float* __restrict__ b2,   // [512]
                const float* __restrict__ rnd,  // [B,C]
                float* __restrict__ fm) {       // [B,C] output map
    __shared__ float sY[16][CC];   // y tile (fp32), reused for sigmoid output
    __shared__ float sH[16][HID];  // hidden activations

    const int tid   = threadIdx.x;
    const int lane  = tid & 31;
    const int wid   = tid >> 5;              // 0..15
    const int rowb  = blockIdx.x * 16;       // first batch row of this tile

    // ---- stage y tile into LDS ----
    for (int i = tid; i < 16 * CC; i += 512) {
        int r = i >> 9, c = i & (CC - 1);
        sY[r][c] = yws[(rowb + r) * CC + c];
    }
    __syncthreads();

    // Lane-layout helpers, contiguous k order for A/B (16-bit A 16x32, 16-bit B 32x16, 32-bit C 16x16)
    const int rowA  = lane & 15;
    const int koffA = (lane >> 4) ? 8 : 0;    // unused: A uses koffB (lanes 16-31 hold K+16)
    const int koffB = (lane >> 4) ? 16 : 0;   // B: lanes 16-31 hold K+16
    const int moff  = (lane >> 4) ? 8 : 0;    // C: lanes 16-31 hold M+8

    // ---- GEMM1: waves 0,1 each own one 16-column tile of hid ----
    if (wid < 2) {
        const int ncol = wid * 16 + (lane & 15);
        v8f acc = {};
        for (int kk = 0; kk < 16; ++kk) {
            const int kb = kk * 32;
            v16h a, b;
#pragma unroll
            for (int j = 0; j < 16; ++j) {
                const int ka = kb + koffB + j;
                a[j] = (_Float16)sY[rowA][ka];
                b[j] = (_Float16)w1[(kb + koffB + j) * HID + ncol];
            }
            acc = __builtin_amdgcn_wmma_f32_16x16x32_f16(
                false, a, false, b, (short)0, acc, false, false);
        }
        const float slope = *pa;
        const float bias  = b1[ncol];
#pragma unroll
        for (int r = 0; r < 8; ++r) {
            float hv = acc[r] + bias;
            hv = (hv >= 0.0f) ? hv : slope * hv;
            sH[moff + r][ncol] = hv;
        }
    }
    __syncthreads();

    // ---- GEMM2: 32 N-tiles over all 16 waves (2 each), K = 32 -> 1 wmma ----
    v16h a2;
#pragma unroll
    for (int j = 0; j < 16; ++j) {
        const int k = koffB + j;
        a2[j] = (_Float16)sH[rowA][k];
    }
#pragma unroll
    for (int t = 0; t < 2; ++t) {
        const int ncol = (wid * 2 + t) * 16 + (lane & 15);
        v16h b;
#pragma unroll
        for (int j = 0; j < 16; ++j)
            b[j] = (_Float16)w2[(koffB + j) * CC + ncol];
        v8f acc = {};
        acc = __builtin_amdgcn_wmma_f32_16x16x32_f16(
            false, a2, false, b, (short)0, acc, false, false);
        const float bias = b2[ncol];
#pragma unroll
        for (int r = 0; r < 8; ++r) {
            const float v = acc[r] + bias;
            sY[moff + r][ncol] = 1.0f / (1.0f + __expf(-v));  // sigmoid
        }
    }
    __syncthreads();

    // ---- top-k threshold by exact rank counting; wave `wid` owns row `wid` ----
    // mask[c] = (y[c] < k-th largest)  <=>  #{j : y[j] > y[c]} >= k  (distinct vals)
    const int row = wid;
    float myv[16];
    int   cnt[16];
#pragma unroll
    for (int i = 0; i < 16; ++i) { myv[i] = sY[row][lane * 16 + i]; cnt[i] = 0; }
    for (int j = 0; j < CC; ++j) {
        const float v = sY[row][j];   // broadcast read
#pragma unroll
        for (int i = 0; i < 16; ++i) cnt[i] += (v > myv[i]) ? 1 : 0;
    }
    const int gb = (rowb + row) * CC;
#pragma unroll
    for (int i = 0; i < 16; ++i) {
        const int   c  = lane * 16 + i;
        const float yv = myv[i];
        const bool below = (cnt[i] >= KSEL);            // strictly below threshold
        const bool rb    = (rnd[gb + c] < 0.5f);
        fm[gb + c] = rb ? (below ? yv : 0.0f) : yv;     // mask*rb*y + y*(1-rb)
    }
}

// ---------------------------------------------------------------------------
// Kernel 3: out = x * final_map[b,c]  (pure streaming, float4, 1 elem/thread)
// ---------------------------------------------------------------------------
__global__ void k_apply(const float4* __restrict__ x4,
                        const float* __restrict__ fm,
                        float4* __restrict__ o4, int n4) {
    const int i = blockIdx.x * blockDim.x + threadIdx.x;
    if (i >= n4) return;
    const int row = i / (HW / 4);          // 784 float4 per (b,c) row
    const float m = fm[row];
    float4 v = x4[i];
    v.x *= m; v.y *= m; v.z *= m; v.w *= m;
    o4[i] = v;
}

// ---------------------------------------------------------------------------
extern "C" void kernel_launch(void* const* d_in, const int* in_sizes, int n_in,
                              void* d_out, int out_size, void* d_ws, size_t ws_size,
                              hipStream_t stream) {
    const float* x    = (const float*)d_in[0];
    const float* w1   = (const float*)d_in[1];
    const float* b1   = (const float*)d_in[2];
    const float* pa   = (const float*)d_in[3];
    const float* w2   = (const float*)d_in[4];
    const float* b2   = (const float*)d_in[5];
    const float* rnd  = (const float*)d_in[6];
    float*       out  = (float*)d_out;

    float* yws = (float*)d_ws;            // [B*C]
    float* fm  = yws + BB * CC;           // [B*C]

    // 1) mean+max reduction: one wave per (b,c) row; 8 waves / block
    {
        const int rows   = BB * CC;               // 16384
        const int blocks = rows / 8;              // 2048
        k_reduce_meanmax<<<blocks, 256, 0, stream>>>(x, yws);
    }
    // 2) fused MLP (WMMA) + sigmoid + top-k + gating
    {
        k_mlp_topk<<<BB / 16, 512, 0, stream>>>(yws, w1, b1, pa, w2, b2, rnd, fm);
    }
    // 3) broadcast scale
    {
        const int n4     = BB * CC * (HW / 4);    // 12,845,056
        const int blocks = (n4 + 255) / 256;      // 50,176
        k_apply<<<blocks, 256, 0, stream>>>((const float4*)x, fm, (float4*)out, n4);
    }
}
